// Encoder_59425167507712
// MI455X (gfx1250) — matrix operand order repaired and checked by reading, not compiled, not run on hardware
//
#include <hip/hip_runtime.h>
#include <hip/hip_bf16.h>

// ---------------------------------------------------------------------------
// CDNA5 (gfx1250) bf16-WMMA transformer encoder
//   D = A[M,K] * B[K,N] + bias, with B stored pre-transposed as Bt[N,K].
//   All matmuls lower to v_wmma_f32_16x16x32_bf16 (wave32).
//   Wave tile 64x64 (4x4 WMMA tiles): 8 fragment loads feed 16 WMMAs
//   -> 32 FLOP/byte at the L1/WGP$ interface.
// ---------------------------------------------------------------------------

typedef __attribute__((ext_vector_type(16))) __bf16 v16bf;
typedef __attribute__((ext_vector_type(8)))  __bf16 v8bf;
typedef __attribute__((ext_vector_type(8)))  float  v8f;

#define LAYERS 4
#define DMODEL 1024
#define BATCH  8
#define SEQ    512
#define DKH    64
#define FFDIM  (4 * DMODEL)
#define MROWS  (BATCH * SEQ)   // 4096 token rows
#define BN_EPS 1e-3f

// ---- WMMA fragment loaders ------------------------------------------------
// A (bf16, row-major, ld = K): 16x32 tile. ISA layout: lane L -> row M = L&15,
// K-halves {kb..kb+7} U {kb+16..kb+23}, kb = 8*(L>=16). Two 16B loads.
__device__ __forceinline__ v16bf load_a_frag(const __bf16* __restrict__ A,
                                             int lda, int m0, int k0, int lane) {
  const int row = m0 + (lane & 15);
  const int kb  = k0 + ((lane >> 4) << 3);
  const v8bf lo = *reinterpret_cast<const v8bf*>(A + (size_t)row * lda + kb);
  const v8bf hi = *reinterpret_cast<const v8bf*>(A + (size_t)row * lda + kb + 16);
  v16bf r;
#pragma unroll
  for (int i = 0; i < 8; ++i) { r[i] = lo[i]; r[i + 8] = hi[i]; }
  return r;
}

// B (bf16) supplied transposed: Bt[N][K], ld = K. 32x16 tile. Lane L -> col
// N = L&15, K-halves {kb..kb+7} U {kb+16..kb+23}, kb = 8*(L>=16). Two 16B loads.
__device__ __forceinline__ v16bf load_b_frag(const __bf16* __restrict__ Bt,
                                             int ldb, int n0, int k0, int lane) {
  const int col = n0 + (lane & 15);
  const int kb  = k0 + ((lane >> 4) << 3);
  const v8bf* p = reinterpret_cast<const v8bf*>(Bt + (size_t)col * ldb + kb);
  const v8bf lo = p[0];
  const v8bf hi = p[2];
  v16bf r;
#pragma unroll
  for (int i = 0; i < 8; ++i) { r[i] = lo[i]; r[i + 8] = hi[i]; }
  return r;
}

// ---- GEMM -----------------------------------------------------------------
// Wave tile = (WM*16) x (WN*16); block = WVM x WVN waves.
// EPI: 0 = f32 out (C = alpha*acc + bias)
//      1 = bf16 out
//      2 = relu -> bf16 out
//      3 = bf16 out transposed per batch-of-Sl rows: out[b][col][row%Sl]
template <int EPI, int WM, int WN, int WVM, int WVN>
__global__ __launch_bounds__(WVM * WVN * 32) void gemm_bf16_wmma(
    const __bf16* __restrict__ A, const __bf16* __restrict__ Bt,
    const float* __restrict__ bias, void* __restrict__ Cout,
    int M, int N, int K,
    long long sA, long long sB, long long sC,   // batch strides (elements)
    float alpha, int Sl) {
  const int bz = blockIdx.z;
  A  += (size_t)bz * sA;
  Bt += (size_t)bz * sB;

  const int w    = threadIdx.x >> 5;
  const int lane = threadIdx.x & 31;
  const int m0   = blockIdx.x * (WM * 16 * WVM) + (w / WVN) * (WM * 16);
  const int n0   = blockIdx.y * (WN * 16 * WVN) + (w % WVN) * (WN * 16);

  v8f acc[WM][WN] = {};

  for (int k0 = 0; k0 < K; k0 += 32) {
    if (k0 + 64 < K) {  // near-scope prefetch two K-steps ahead
      __builtin_prefetch(A  + (size_t)(m0 + (lane & 15)) * K + (k0 + 64), 0, 3);
      __builtin_prefetch(Bt + (size_t)(n0 + (lane & 15)) * K + (k0 + 64), 0, 3);
    }
    v16bf af[WM];
    v16bf bf[WN];
#pragma unroll
    for (int i = 0; i < WM; ++i) af[i] = load_a_frag(A, K, m0 + i * 16, k0, lane);
#pragma unroll
    for (int j = 0; j < WN; ++j) bf[j] = load_b_frag(Bt, K, n0 + j * 16, k0, lane);
#pragma unroll
    for (int i = 0; i < WM; ++i)
#pragma unroll
      for (int j = 0; j < WN; ++j)
        acc[i][j] = __builtin_amdgcn_wmma_f32_16x16x32_bf16(
            false, af[i], false, bf[j], (short)0, acc[i][j], false, false);
  }

  // C/D layout: VGPR r -> row tile_m + 8*(lane>=16) + r, col = tile_n + (lane&15)
  const int colL = lane & 15;
  const int rbL  = (lane >> 4) << 3;
#pragma unroll
  for (int i = 0; i < WM; ++i) {
#pragma unroll
    for (int j = 0; j < WN; ++j) {
      const int col = n0 + j * 16 + colL;
      const int rb  = m0 + i * 16 + rbL;
      const float bval = (bias != nullptr) ? bias[col] : 0.0f;
#pragma unroll
      for (int r = 0; r < 8; ++r) {
        const int row = rb + r;
        const float v = acc[i][j][r] * alpha + bval;
        if (EPI == 0) {
          ((float*)Cout)[(size_t)bz * sC + (size_t)row * N + col] = v;
        } else if (EPI == 1) {
          ((__bf16*)Cout)[(size_t)bz * sC + (size_t)row * N + col] = (__bf16)v;
        } else if (EPI == 2) {
          ((__bf16*)Cout)[(size_t)bz * sC + (size_t)row * N + col] = (__bf16)fmaxf(v, 0.0f);
        } else {  // EPI == 3: per-Sl-batch transposed store (builds V^T)
          const int bb = row / Sl;
          const int rr = row - bb * Sl;
          ((__bf16*)Cout)[((size_t)bb * N + col) * Sl + rr] = (__bf16)v;
        }
      }
    }
  }
}

// ---- embedding gather + pes[s,0] scalar add (reference bug preserved) -----
__global__ __launch_bounds__(256) void embed_kernel(
    const int* __restrict__ seq, const float* __restrict__ emb,
    const float* __restrict__ pes, float* __restrict__ xf,
    __bf16* __restrict__ xb, int total, int D, int S) {
  const int idx = blockIdx.x * blockDim.x + threadIdx.x;
  if (idx >= total) return;
  const int i = idx / D;
  const int d = idx - i * D;
  const int s = i % S;
  const float v = emb[(size_t)seq[i] * D + d] + pes[(size_t)s * D];  // pes[s,0]
  xf[idx] = v;
  xb[idx] = (__bf16)v;
}

// ---- residual add + inference BatchNorm; writes f32 + bf16 activation -----
__global__ __launch_bounds__(256) void add_bn_kernel(
    const float* __restrict__ x, const float* __restrict__ y,
    const float* __restrict__ gamma, const float* __restrict__ beta,
    const float* __restrict__ mean, const float* __restrict__ var,
    float* __restrict__ xo, __bf16* __restrict__ xo16, int total, int D) {
  const int idx = blockIdx.x * blockDim.x + threadIdx.x;
  if (idx >= total) return;
  const int d = idx % D;
  const float v = (x[idx] + y[idx] - mean[d]) * rsqrtf(var[d] + BN_EPS) * gamma[d] + beta[d];
  xo[idx] = v;
  xo16[idx] = (__bf16)v;
}

// ---- wave-per-row softmax, f32 in -> bf16 probs out -----------------------
__global__ __launch_bounds__(256) void softmax_bf16_kernel(
    const float* __restrict__ S_, __bf16* __restrict__ P, int rows, int n) {
  const int wave = (blockIdx.x * blockDim.x + threadIdx.x) >> 5;
  const int lane = threadIdx.x & 31;
  if (wave >= rows) return;
  const float* src = S_ + (size_t)wave * n;
  float m = -3.4e38f;
  for (int j = lane; j < n; j += 32) m = fmaxf(m, src[j]);
#pragma unroll
  for (int off = 16; off; off >>= 1) m = fmaxf(m, __shfl_xor(m, off, 32));
  float s = 0.0f;
  for (int j = lane; j < n; j += 32) s += __expf(src[j] - m);
#pragma unroll
  for (int off = 16; off; off >>= 1) s += __shfl_xor(s, off, 32);
  const float inv = 1.0f / s;
  __bf16* dst = P + (size_t)wave * n;
  for (int j = lane; j < n; j += 32) dst[j] = (__bf16)(__expf(src[j] - m) * inv);
}

// ---- weight pack: f32 W[K][N] -> bf16 Wt[N][K] ----------------------------
__global__ __launch_bounds__(256) void packT_kernel(
    const float* __restrict__ W, __bf16* __restrict__ Wt, int K, int N) {
  const int idx = blockIdx.x * blockDim.x + threadIdx.x;
  if (idx >= K * N) return;
  const int k = idx / N;
  const int n = idx - k * N;
  Wt[(size_t)n * K + k] = (__bf16)W[idx];
}

// ---------------------------------------------------------------------------
extern "C" void kernel_launch(void* const* d_in, const int* in_sizes, int n_in,
                              void* d_out, int out_size, void* d_ws, size_t ws_size,
                              hipStream_t stream) {
  (void)in_sizes; (void)n_in; (void)out_size; (void)ws_size;

  const int*   seq = (const int*)  d_in[0];
  const float* emb = (const float*)d_in[1];
  const float* pes = (const float*)d_in[2];
  const float* wq  = (const float*)d_in[3];
  const float* bq  = (const float*)d_in[4];
  const float* wk  = (const float*)d_in[5];
  const float* bk  = (const float*)d_in[6];
  const float* wv  = (const float*)d_in[7];
  const float* bv  = (const float*)d_in[8];
  const float* wo  = (const float*)d_in[9];
  const float* bo  = (const float*)d_in[10];
  const float* ag  = (const float*)d_in[11];
  const float* ab  = (const float*)d_in[12];
  const float* am  = (const float*)d_in[13];
  const float* av  = (const float*)d_in[14];
  const float* w1  = (const float*)d_in[15];
  const float* b1  = (const float*)d_in[16];
  const float* w2  = (const float*)d_in[17];
  const float* b2  = (const float*)d_in[18];
  const float* fg  = (const float*)d_in[19];
  const float* fb  = (const float*)d_in[20];
  const float* fm  = (const float*)d_in[21];
  const float* fv  = (const float*)d_in[22];

  // ---- bump-allocate workspace (≈152 MB) ----
  char* base = (char*)d_ws;
  size_t off = 0;
  auto alloc = [&](size_t bytes) -> void* {
    void* p = base + off;
    off += (bytes + 255) & ~(size_t)255;
    return p;
  };
  float*  xf  = (float*) alloc((size_t)MROWS * DMODEL * 4);      // activation f32
  __bf16* xb  = (__bf16*)alloc((size_t)MROWS * DMODEL * 2);      // activation bf16
  __bf16* qb  = (__bf16*)alloc((size_t)MROWS * DKH * 2);
  __bf16* kb  = (__bf16*)alloc((size_t)MROWS * DKH * 2);
  __bf16* vtb = (__bf16*)alloc((size_t)BATCH * DKH * SEQ * 2);   // V^T [b][dk][s]
  __bf16* hb  = (__bf16*)alloc((size_t)MROWS * DKH * 2);         // attn head
  float*  sc  = (float*) alloc((size_t)BATCH * SEQ * SEQ * 4);   // scores
  __bf16* pb  = (__bf16*)alloc((size_t)BATCH * SEQ * SEQ * 2);   // probs
  float*  yf  = (float*) alloc((size_t)MROWS * DMODEL * 4);      // att / ffn out
  __bf16* h1b = (__bf16*)alloc((size_t)MROWS * FFDIM * 2);       // relu hidden
  __bf16* wqT = (__bf16*)alloc((size_t)LAYERS * DMODEL * DKH * 2);
  __bf16* wkT = (__bf16*)alloc((size_t)LAYERS * DMODEL * DKH * 2);
  __bf16* wvT = (__bf16*)alloc((size_t)LAYERS * DMODEL * DKH * 2);
  __bf16* woT = (__bf16*)alloc((size_t)LAYERS * DKH * DMODEL * 2);
  __bf16* w1T = (__bf16*)alloc((size_t)LAYERS * DMODEL * FFDIM * 2);
  __bf16* w2T = (__bf16*)alloc((size_t)LAYERS * FFDIM * DMODEL * 2);

  const dim3 blk256(256);
  const dim3 blk128(128);
  auto g1 = [](size_t n) { return dim3((unsigned)((n + 255) / 256)); };

  // ---- pack all weights to bf16 transposed [N][K] (one-time, mem-bound) ----
  for (int l = 0; l < LAYERS; ++l) {
    packT_kernel<<<g1((size_t)DMODEL * DKH), blk256, 0, stream>>>(wq + (size_t)l * DMODEL * DKH, wqT + (size_t)l * DMODEL * DKH, DMODEL, DKH);
    packT_kernel<<<g1((size_t)DMODEL * DKH), blk256, 0, stream>>>(wk + (size_t)l * DMODEL * DKH, wkT + (size_t)l * DMODEL * DKH, DMODEL, DKH);
    packT_kernel<<<g1((size_t)DMODEL * DKH), blk256, 0, stream>>>(wv + (size_t)l * DMODEL * DKH, wvT + (size_t)l * DMODEL * DKH, DMODEL, DKH);
    packT_kernel<<<g1((size_t)DKH * DMODEL), blk256, 0, stream>>>(wo + (size_t)l * DKH * DMODEL, woT + (size_t)l * DKH * DMODEL, DKH, DMODEL);
    packT_kernel<<<g1((size_t)DMODEL * FFDIM), blk256, 0, stream>>>(w1 + (size_t)l * DMODEL * FFDIM, w1T + (size_t)l * DMODEL * FFDIM, DMODEL, FFDIM);
    packT_kernel<<<g1((size_t)FFDIM * DMODEL), blk256, 0, stream>>>(w2 + (size_t)l * FFDIM * DMODEL, w2T + (size_t)l * FFDIM * DMODEL, FFDIM, DMODEL);
  }

  // ---- embedding + positional scalar ----
  embed_kernel<<<g1((size_t)MROWS * DMODEL), blk256, 0, stream>>>(
      seq, emb, pes, xf, xb, MROWS * DMODEL, DMODEL, SEQ);

  const float inv_sqrt_dk = 0.125f;  // 1/sqrt(64)

  for (int l = 0; l < LAYERS; ++l) {
    const __bf16* wqTl = wqT + (size_t)l * DMODEL * DKH;
    const __bf16* wkTl = wkT + (size_t)l * DMODEL * DKH;
    const __bf16* wvTl = wvT + (size_t)l * DMODEL * DKH;
    const __bf16* woTl = woT + (size_t)l * DKH * DMODEL;
    const __bf16* w1Tl = w1T + (size_t)l * DMODEL * FFDIM;
    const __bf16* w2Tl = w2T + (size_t)l * FFDIM * DMODEL;

    // Q, K projections -> bf16 [MROWS, DK]  (tall config: 256x64 block)
    gemm_bf16_wmma<1, 4, 4, 4, 1><<<dim3(MROWS / 256, 1, 1), blk128, 0, stream>>>(
        xb, wqTl, bq + (size_t)l * DKH, qb, MROWS, DKH, DMODEL, 0, 0, 0, 1.0f, 0);
    gemm_bf16_wmma<1, 4, 4, 4, 1><<<dim3(MROWS / 256, 1, 1), blk128, 0, stream>>>(
        xb, wkTl, bk + (size_t)l * DKH, kb, MROWS, DKH, DMODEL, 0, 0, 0, 1.0f, 0);
    // V projection, stored transposed per batch: vtb[b][dk][s]
    gemm_bf16_wmma<3, 4, 4, 4, 1><<<dim3(MROWS / 256, 1, 1), blk128, 0, stream>>>(
        xb, wvTl, bv + (size_t)l * DKH, vtb, MROWS, DKH, DMODEL, 0, 0, 0, 1.0f, SEQ);

    // scores = (Q K^T) / sqrt(dk), batched over B (K operand already [S][DK])
    gemm_bf16_wmma<0, 4, 4, 2, 2><<<dim3(SEQ / 128, SEQ / 128, BATCH), blk128, 0, stream>>>(
        qb, kb, nullptr, sc, SEQ, SEQ, DKH,
        (long long)SEQ * DKH, (long long)SEQ * DKH, (long long)SEQ * SEQ,
        inv_sqrt_dk, 0);

    softmax_bf16_kernel<<<g1((size_t)BATCH * SEQ * 32), blk256, 0, stream>>>(
        sc, pb, BATCH * SEQ, SEQ);

    // head = probs @ V  (Bt = V^T[b], ld = SEQ), batched (tall config)
    gemm_bf16_wmma<1, 4, 4, 4, 1><<<dim3(SEQ / 256, 1, BATCH), blk128, 0, stream>>>(
        pb, vtb, nullptr, hb, SEQ, DKH, SEQ,
        (long long)SEQ * SEQ, (long long)DKH * SEQ, (long long)SEQ * DKH,
        1.0f, 0);

    // att = head @ Wo + bo -> f32
    gemm_bf16_wmma<0, 4, 4, 2, 2><<<dim3(MROWS / 128, DMODEL / 128, 1), blk128, 0, stream>>>(
        hb, woTl, bo + (size_t)l * DMODEL, yf, MROWS, DMODEL, DKH, 0, 0, 0, 1.0f, 0);

    // x = BN(x + att)
    add_bn_kernel<<<g1((size_t)MROWS * DMODEL), blk256, 0, stream>>>(
        xf, yf, ag + (size_t)l * DMODEL, ab + (size_t)l * DMODEL,
        am + (size_t)l * DMODEL, av + (size_t)l * DMODEL,
        xf, xb, MROWS * DMODEL, DMODEL);

    // h1 = relu(x @ W1 + b1) -> bf16  (dominant GEMM)
    gemm_bf16_wmma<2, 4, 4, 2, 2><<<dim3(MROWS / 128, FFDIM / 128, 1), blk128, 0, stream>>>(
        xb, w1Tl, b1 + (size_t)l * FFDIM, h1b, MROWS, FFDIM, DMODEL, 0, 0, 0, 1.0f, 0);

    // ffn = h1 @ W2 + b2 -> f32  (dominant GEMM)
    gemm_bf16_wmma<0, 4, 4, 2, 2><<<dim3(MROWS / 128, DMODEL / 128, 1), blk128, 0, stream>>>(
        h1b, w2Tl, b2 + (size_t)l * DMODEL, yf, MROWS, DMODEL, FFDIM, 0, 0, 0, 1.0f, 0);

    // x = BN(x + ffn); final layer writes straight to d_out
    float* xdst = (l == LAYERS - 1) ? (float*)d_out : xf;
    add_bn_kernel<<<g1((size_t)MROWS * DMODEL), blk256, 0, stream>>>(
        xf, yf, fg + (size_t)l * DMODEL, fb + (size_t)l * DMODEL,
        fm + (size_t)l * DMODEL, fv + (size_t)l * DMODEL,
        xdst, xb, MROWS * DMODEL, DMODEL);
  }
}
